// DisentangledSelfAttention_1434519077097
// MI455X (gfx1250) — matrix operand order repaired and checked by reading, not compiled, not run on hardware
//
#include <hip/hip_runtime.h>
#include <hip/hip_bf16.h>

// ---------------------------------------------------------------------------
// DisentangledSelfAttention for MI455X (gfx1250, wave32, WMMA bf16, async LDS)
//
// B=2048, F=64, D=A=512, H=8, HD=64.
// The reference's "unary" term softmaxes over a size-1 axis -> exactly 1.0,
// so K2/Wk2/mu_K2 never influence the output:
//   out = relu((softmax(Qc Kc^T) + 1) @ V + query)     (3 GEMMs, not 4)
//
// Stage 1: X@W projections. A 128-row strip of X is staged once in LDS
//          (bf16, 133KB) and reused across all 8 N-blocks -> X is read from
//          HBM exactly once (~1.2GB total traffic ~= 52us at 23.3 TB/s).
//          Uses 207KB LDS per workgroup (CDNA5 WGP has 320KB).
// Stage 2: per-(b,h) attention; head slices staged via
//          GLOBAL_LOAD_ASYNC_TO_LDS_B128 (ASYNCcnt), WMMA for S and P@V.
// ---------------------------------------------------------------------------

typedef __attribute__((ext_vector_type(16))) __bf16 v16bf;
typedef __attribute__((ext_vector_type(8)))  __bf16 v8bf;
typedef __attribute__((ext_vector_type(8)))  float  v8f;

#define BB 2048
#define FF 64
#define DD 512
#define AA 512
#define HH 8
#define HD 64

#define ASTR 520                       // A-strip LDS row stride (bf16 elems)
#define BSTR 72                        // B-strip LDS row stride (bf16 elems)

static __device__ __forceinline__ v8f wmma_bf16(v16bf a, v16bf b, v8f c) {
    return __builtin_amdgcn_wmma_f32_16x16x32_bf16(
        /*neg_a=*/false, a, /*neg_b=*/false, b,
        /*c_mod=*/(short)0, c, /*reuse_a=*/false, /*reuse_b=*/false);
}

// A-operand fragment (16 rows x 32 k, bf16). Per ISA 7.12.2:
//   lanes 0-15 : row M = lane,    elems = K{0..7, 16..23}
//   lanes 16-31: row M = lane-16, elems = K{8..15, 24..31}
static __device__ __forceinline__ v16bf load_frag_A(const __bf16* base, int stride, int lane) {
    const int m  = lane & 15;
    const int k0 = (lane < 16) ? 0 : 8;
    const __bf16* r = base + m * stride;
    union { v16bf v; v8bf h[2]; } u;
    u.h[0] = *(const v8bf*)(r + k0);
    u.h[1] = *(const v8bf*)(r + k0 + 16);
    return u.v;
}

// B-operand fragment (32 k x 16 cols, bf16):
//   lanes 0-15 : col N = lane,    elems = K{0..7, 16..23}
//   lanes 16-31: col N = lane-16, elems = K{8..15, 24..31}
static __device__ __forceinline__ v16bf load_frag_B(const __bf16* base, int stride, int lane) {
    const int n  = lane & 15;
    const int k0 = (lane < 16) ? 0 : 8;
    v16bf v;
    #pragma unroll
    for (int s = 0; s < 8; ++s) {
        v[s]     = base[(k0 + s) * stride + n];
        v[s + 8] = base[(k0 + 16 + s) * stride + n];
    }
    return v;
}

// Async 16-byte global -> LDS copy (CDNA5 TDM-family path, ASYNCcnt-tracked).
// The low 32 bits of a flat pointer to LDS are the in-group LDS byte offset.
static __device__ __forceinline__ void async_copy_b128(void* lds_dst, const void* gsrc) {
    const unsigned lds_off = (unsigned)(unsigned long long)(uintptr_t)lds_dst;
    const unsigned long long ga = (unsigned long long)(uintptr_t)gsrc;
    asm volatile("global_load_async_to_lds_b128 %0, %1, off"
                 :: "v"(lds_off), "v"(ga) : "memory");
}
static __device__ __forceinline__ void async_wait0() {
    asm volatile("s_wait_asynccnt 0" ::: "memory");
}

// ---------------------------------------------------------------------------
// Kernel A: projection GEMM.  out[g] = X[g] @ W[g] + b[g], bf16 result.
// grid = (M/128, 3), block = 256 (8 waves: 4 along M, 2 along N).
// Dynamic LDS: A strip 128x520 bf16 + B strip 512x72 bf16 = 206,848 B.
// ---------------------------------------------------------------------------
__global__ __launch_bounds__(256)
void proj_gemm_kernel(const float* __restrict__ xq,
                      const float* __restrict__ xk,
                      const float* __restrict__ xv,
                      const float* __restrict__ wq,
                      const float* __restrict__ wk,
                      const float* __restrict__ wv,
                      const float* __restrict__ bq,
                      const float* __restrict__ bk,
                      const float* __restrict__ bv,
                      __bf16* __restrict__ ws) {
    extern __shared__ __bf16 smem[];
    __bf16* As = smem;                    // [128][ASTR]
    __bf16* Bs = smem + 128 * ASTR;       // [512][BSTR]

    const int g = blockIdx.y;
    const float* X  = (g == 0) ? xq : (g == 1) ? xk : xv;
    const float* W  = (g == 0) ? wq : (g == 1) ? wk : wv;
    const float* Bi = (g == 0) ? bq : (g == 1) ? bk : bv;
    const size_t SZ = (size_t)BB * FF * AA;
    __bf16* out = ws + (size_t)g * SZ;

    const int t     = threadIdx.x;
    const int lane  = t & 31;
    const int wid   = t >> 5;
    const int waveM = wid & 3;            // 0..3 -> 32-row strip
    const int waveN = wid >> 2;           // 0..1 -> 32-col strip
    const int rowBase = blockIdx.x * 128;

    // ---- stage full A strip once: 128x512 f32 -> bf16 (64 float4/thread) ----
    #pragma unroll 8
    for (int i = 0; i < 64; ++i) {
        const int chunk = t + i * 256;    // 0..16383
        const int r  = chunk >> 7;        // 128 float4 per row
        const int c4 = chunk & 127;
        const float4 x4 = *(const float4*)(X + (size_t)(rowBase + r) * DD + c4 * 4);
        union { __bf16 b[4]; uint2 u; } cv;
        cv.b[0] = (__bf16)x4.x; cv.b[1] = (__bf16)x4.y;
        cv.b[2] = (__bf16)x4.z; cv.b[3] = (__bf16)x4.w;
        *(uint2*)(As + r * ASTR + c4 * 4) = cv.u;
    }

    // ---- loop N-blocks locally; X never re-read from HBM ----
    for (int nb = 0; nb < AA / 64; ++nb) {
        __syncthreads();                  // protect previous B strip reads
        // stage B strip: 512x64 f32 -> bf16 (32 float4/thread; L2-resident W)
        #pragma unroll 8
        for (int i = 0; i < 32; ++i) {
            const int chunk = t + i * 256;      // 0..8191
            const int r  = chunk >> 4;          // 16 float4 per row
            const int c4 = chunk & 15;
            const float4 w4 = *(const float4*)(W + (size_t)r * AA + nb * 64 + c4 * 4);
            union { __bf16 b[4]; uint2 u; } cv;
            cv.b[0] = (__bf16)w4.x; cv.b[1] = (__bf16)w4.y;
            cv.b[2] = (__bf16)w4.z; cv.b[3] = (__bf16)w4.w;
            *(uint2*)(Bs + r * BSTR + c4 * 4) = cv.u;
        }
        __syncthreads();

        v8f acc[2][2] = {};
        #pragma unroll 4
        for (int kk = 0; kk < 16; ++kk) {       // K = 16 x 32
            v16bf af[2], bf[2];
            #pragma unroll
            for (int mi = 0; mi < 2; ++mi)
                af[mi] = load_frag_A(As + (waveM * 32 + mi * 16) * ASTR + kk * 32,
                                     ASTR, lane);
            #pragma unroll
            for (int ni = 0; ni < 2; ++ni)
                bf[ni] = load_frag_B(Bs + (kk * 32) * BSTR + waveN * 32 + ni * 16,
                                     BSTR, lane);
            #pragma unroll
            for (int mi = 0; mi < 2; ++mi)
                #pragma unroll
                for (int ni = 0; ni < 2; ++ni)
                    acc[mi][ni] = wmma_bf16(af[mi], bf[ni], acc[mi][ni]);
        }

        // ---- epilogue: + bias, f32 -> bf16 store ----
        // C/D layout: elem v at lane l -> row = v + (l<16?0:8), col = l&15
        const int rofs = (lane < 16) ? 0 : 8;
        #pragma unroll
        for (int ni = 0; ni < 2; ++ni) {
            const int col = nb * 64 + waveN * 32 + ni * 16 + (lane & 15);
            const float bv_ = Bi[col];
            #pragma unroll
            for (int mi = 0; mi < 2; ++mi) {
                #pragma unroll
                for (int v = 0; v < 8; ++v) {
                    const int row = rowBase + waveM * 32 + mi * 16 + v + rofs;
                    out[(size_t)row * AA + col] = (__bf16)(acc[mi][ni][v] + bv_);
                }
            }
        }
    }
}

// ---------------------------------------------------------------------------
// Kernel B: per-(b,h) attention.  grid = (B, H), block = 128 (4 waves).
// ---------------------------------------------------------------------------
#define HROW 72                       // padded LDS row stride (bf16 elems)
#define HSZ  (64 * HROW)              // one 64x64 head slice (padded)

__global__ __launch_bounds__(128)
void attn_kernel(const __bf16* __restrict__ ws,
                 const float* __restrict__ query,
                 float* __restrict__ out) {
    // One fused staging buffer: [0) Q  [HSZ) K  [2*HSZ) V
    __shared__ __bf16 qkvS[3 * HSZ];
    __shared__ __bf16 kcT [HSZ];      // centered K, transposed: kcT[c][f]
    __shared__ __bf16 ps  [HSZ];      // softmax(S)+1 as bf16
    __shared__ float  sS  [64 * 68];  // raw scores fp32
    __shared__ float  meanQ[64], meanK[64];

    __bf16* qs = qkvS;
    __bf16* ks = qkvS + HSZ;
    __bf16* vs = qkvS + 2 * HSZ;

    const int b = blockIdx.x;
    const int h = blockIdx.y;
    const int t = threadIdx.x;
    const int lane = t & 31;
    const int w = t >> 5;             // wave id 0..3 -> 16-row strip
    const size_t SZ = (size_t)BB * FF * AA;
    const size_t headBase = (size_t)b * FF * AA + (size_t)h * HD;

    // ---- stage Q,K,V head slices via async global->LDS b128 copies ----
    // 3 tensors x 64 rows x 8 chunks(16B) = 1536 copies / 128 threads = 12 each
    #pragma unroll
    for (int i = 0; i < 12; ++i) {
        const int id  = t + i * 128;          // 0..1535
        const int ten = id >> 9;              // 0:Q 1:K 2:V
        const int rem = id & 511;
        const int f   = rem >> 3;
        const int c8  = rem & 7;
        async_copy_b128(qkvS + ten * HSZ + f * HROW + c8 * 8,
                        ws + (size_t)ten * SZ + headBase + (size_t)f * AA + c8 * 8);
    }
    async_wait0();                    // drain this wave's ASYNCcnt
    __syncthreads();

    // ---- column means over the field axis ----
    if (t < 64) {
        float sq = 0.f, sk = 0.f;
        #pragma unroll 4
        for (int f = 0; f < FF; ++f) {
            sq += (float)qs[f * HROW + t];
            sk += (float)ks[f * HROW + t];
        }
        meanQ[t] = sq * (1.0f / FF);
        meanK[t] = sk * (1.0f / FF);
    }
    __syncthreads();

    // ---- center Q in place; build centered-transposed K ----
    {
        const int f  = t >> 1;
        const int c0 = (t & 1) * 32;
        #pragma unroll 8
        for (int j = 0; j < 32; ++j) {
            const int c = c0 + j;
            qs[f * HROW + c] = (__bf16)((float)qs[f * HROW + c] - meanQ[c]);
            kcT[c * HROW + f] = (__bf16)((float)ks[f * HROW + c] - meanK[c]);
        }
    }
    __syncthreads();

    // ---- S = Qc @ Kc^T  (16x64 strip per wave) ----
    {
        v8f sacc[4] = {};
        #pragma unroll
        for (int ko = 0; ko < HD; ko += 32) {
            const v16bf af = load_frag_A(qs + (w * 16) * HROW + ko, HROW, lane);
            #pragma unroll
            for (int ni = 0; ni < 4; ++ni) {
                const v16bf bfg = load_frag_B(kcT + ko * HROW + ni * 16, HROW, lane);
                sacc[ni] = wmma_bf16(af, bfg, sacc[ni]);
            }
        }
        const int rofs = (lane < 16) ? 0 : 8;
        #pragma unroll
        for (int ni = 0; ni < 4; ++ni)
            #pragma unroll
            for (int v = 0; v < 8; ++v)
                sS[(w * 16 + v + rofs) * 68 + ni * 16 + (lane & 15)] = sacc[ni][v];
    }
    __syncthreads();

    // ---- row softmax over keys, +1.0 (degenerate unary term), -> bf16 ----
    if (t < 64) {
        const float* r = sS + t * 68;
        float m = r[0];
        #pragma unroll 8
        for (int k = 1; k < FF; ++k) m = fmaxf(m, r[k]);
        float s = 0.f;
        #pragma unroll 8
        for (int k = 0; k < FF; ++k) s += __expf(r[k] - m);
        const float inv = 1.0f / s;
        #pragma unroll 8
        for (int k = 0; k < FF; ++k)
            ps[t * HROW + k] = (__bf16)(__expf(r[k] - m) * inv + 1.0f);
    }
    __syncthreads();

    // ---- O = P @ V ; fused residual + relu epilogue (fp32) ----
    {
        v8f oacc[4] = {};
        #pragma unroll
        for (int ko = 0; ko < FF; ko += 32) {
            const v16bf af = load_frag_A(ps + (w * 16) * HROW + ko, HROW, lane);
            #pragma unroll
            for (int ni = 0; ni < 4; ++ni) {
                const v16bf bfg = load_frag_B(vs + ko * HROW + ni * 16, HROW, lane);
                oacc[ni] = wmma_bf16(af, bfg, oacc[ni]);
            }
        }
        const int rofs = (lane < 16) ? 0 : 8;
        #pragma unroll
        for (int ni = 0; ni < 4; ++ni) {
            const int c = ni * 16 + (lane & 15);
            #pragma unroll
            for (int v = 0; v < 8; ++v) {
                const int f = w * 16 + v + rofs;
                const size_t idx = headBase + (size_t)f * AA + c;
                const float r = oacc[ni][v] + query[idx];
                out[idx] = fmaxf(r, 0.0f);
            }
        }
    }
}

// ---------------------------------------------------------------------------
extern "C" void kernel_launch(void* const* d_in, const int* in_sizes, int n_in,
                              void* d_out, int out_size, void* d_ws, size_t ws_size,
                              hipStream_t stream) {
    const float* query = (const float*)d_in[0];
    const float* key   = (const float*)d_in[1];
    const float* value = (const float*)d_in[2];
    const float* Wq    = (const float*)d_in[3];
    const float* bq    = (const float*)d_in[4];
    const float* Wk    = (const float*)d_in[5];
    const float* bk    = (const float*)d_in[6];
    const float* Wv    = (const float*)d_in[7];
    const float* bv    = (const float*)d_in[8];
    // d_in[9]/d_in[10] (Wk2/bk2) feed only the degenerate unary softmax -> unused.
    float* out = (float*)d_out;
    __bf16* ws = (__bf16*)d_ws;   // 3 x [B*F, A] bf16 projection buffers

    // Stage 1: Q/K/V projections (M=131072, N=512, K=512) in bf16 WMMA.
    // Dynamic LDS: 128x520 + 512x72 bf16 = 206,848 B (one WG per 320KB WGP).
    const size_t ldsA = (size_t)(128 * ASTR + 512 * BSTR) * sizeof(__bf16);
    dim3 gA((BB * FF) / 128, 3, 1);
    proj_gemm_kernel<<<gA, 256, ldsA, stream>>>(query, key, value,
                                                Wq, Wk, Wv, bq, bk, bv, ws);

    // Stage 2: per-(b,h) attention with fused softmax/residual/relu.
    dim3 gB(BB, HH, 1);
    attn_kernel<<<gB, 128, 0, stream>>>(ws, query, out);
}
